// AASIST_61392262529421
// MI455X (gfx1250) — matrix operand order repaired and checked by reading, not compiled, not run on hardware
//
#include <hip/hip_runtime.h>
#include <hip/hip_bf16.h>

typedef __attribute__((ext_vector_type(16))) _Float16 v16h;
typedef __attribute__((ext_vector_type(8)))  _Float16 v8h;
typedef __attribute__((ext_vector_type(8)))  float    v8f;

#define SHUF16(lo, hi) __builtin_shufflevector(lo, hi, 0,1,2,3,4,5,6,7,8,9,10,11,12,13,14,15)

#define BB    8
#define TT    64000
#define HH    513          // freq bins
#define WW    251          // frames
#define P2    (HH*WW)      // pixels per batch = 128763
#define NODES 250
#define NDIM  4116
#define KPAD1 4128         // NDIM rounded up to 32
#define MPAD  2048         // BB*NODES rounded up to 64
#define EPSV  1e-5f

__device__ __forceinline__ v8f wmma16x16x32(v16h a, v16h b, v8f c) {
    return __builtin_amdgcn_wmma_f32_16x16x32_f16(false, a, false, b, (short)0, c, false, false);
}

// ---------------------------------------------------------------- FFT + logmag
// one block (128 threads) per (frame, batch). 1024-pt radix-2 DIT in LDS.
__global__ void k_fft_logmag(const float* __restrict__ x, float* __restrict__ sout) {
    __shared__ float re[1024], im[1024];
    int f = blockIdx.x, b = blockIdx.y, tid = threadIdx.x;
    for (int q = 0; q < 8; ++q) {
        int i = tid + q * 128;
        int j = f * 256 + i - 512;               // reflect pad (no edge repeat)
        if (j < 0) j = -j;
        if (j >= TT) j = 2 * TT - 2 - j;
        float v = x[b * TT + j];
        int rb = __brev((unsigned)i) >> 22;      // reverse 10 bits
        re[rb] = v; im[rb] = 0.f;
    }
    __syncthreads();
    for (int s = 0; s < 10; ++s) {
        int half = 1 << s;
        for (int t = tid; t < 512; t += 128) {
            int pos = t & (half - 1);
            int i0  = ((t >> s) << (s + 1)) + pos;
            int i1  = i0 + half;
            float ang = -3.14159265358979f * (float)pos / (float)half;
            float sn, cs; __sincosf(ang, &sn, &cs);
            float tr = cs * re[i1] - sn * im[i1];
            float ti = cs * im[i1] + sn * re[i1];
            re[i1] = re[i0] - tr; im[i1] = im[i0] - ti;
            re[i0] += tr;         im[i0] += ti;
        }
        __syncthreads();
    }
    for (int k = tid; k < HH; k += 128) {
        float mag = sqrtf(re[k] * re[k] + im[k] * im[k]);
        sout[(b * HH + k) * WW + f] = __logf(mag + 1e-9f);
    }
}

// ---------------------------------------------------------------- sinc -> nodesh[:, 0:20] (f16)
__global__ void k_sinc_nodes(const float* __restrict__ x, const float* __restrict__ w,
                             _Float16* __restrict__ nodesh) {
    int id = blockIdx.x * 256 + threadIdx.x;
    if (id >= BB * NODES * 20) return;
    int c = id % 20, i = (id / 20) % NODES, b = id / (20 * NODES);
    float acc = 0.f;
    for (int j = 0; j < 4; ++j) {
        int t0 = i * 4 + j - 512;
        for (int k = 0; k < 1024; ++k) {
            int t = t0 + k;
            bool ok = (t >= 0) & (t < TT);
            float xv = x[b * TT + (ok ? t : 0)];
            acc += (ok ? xv : 0.f) * w[c * 1024 + k];
        }
    }
    nodesh[((size_t)(b * NODES + i)) * KPAD1 + c] = (_Float16)(acc * 0.25f);
}

// ---------------------------------------------------------------- conv1 + BN + relu (NHWC f16 out)
__global__ void k_conv1(const float* __restrict__ s, const float* __restrict__ w1,
                        const float* __restrict__ cb, const float* __restrict__ g,
                        const float* __restrict__ bb, _Float16* __restrict__ h1) {
    __shared__ float wl[288], cbl[32], scl[32], shl[32];
    int tid = threadIdx.x, b = blockIdx.y;
    if (tid < 288) wl[tid] = w1[tid];
    if (tid < 32) { cbl[tid] = cb[tid]; scl[tid] = g[tid] * rsqrtf(1.f + EPSV); shl[tid] = bb[tid]; }
    __syncthreads();
    int p = blockIdx.x * 256 + tid;
    if (p >= P2) return;
    int y = p / WW, xx = p % WW;
    float patch[9];
#pragma unroll
    for (int t = 0; t < 9; ++t) {
        int yy = y + t / 3 - 1, xc = xx + t % 3 - 1;
        bool ok = (yy >= 0) & (yy < HH) & (xc >= 0) & (xc < WW);
        int yyc = ok ? yy : y, xcc = ok ? xc : xx;         // clamp to a valid addr, select 0
        float v = s[(b * HH + yyc) * WW + xcc];
        patch[t] = ok ? v : 0.f;
    }
    v8h* dst = (v8h*)(h1 + (((size_t)b * P2 + p) << 5));
#pragma unroll
    for (int grp = 0; grp < 4; ++grp) {
        v8h pack;
#pragma unroll
        for (int j = 0; j < 8; ++j) {
            int oc = grp * 8 + j;
            float a = cbl[oc];
#pragma unroll
            for (int t = 0; t < 9; ++t) a += wl[oc * 9 + t] * patch[t];
            a = a * scl[oc] + shl[oc];
            pack[j] = (_Float16)(a > 0.f ? a : 0.f);
        }
        dst[grp] = pack;
    }
}

// ---------------------------------------------------------------- conv2 implicit-GEMM WMMA
// block: 256 thr = 8 waves; tile 64 pixels x 32 outch; K = 9 taps x 32 inch.
// Staging is branch-free: clamped b128 load + cndmask-zero + b128 ds store.
__global__ void k_conv2_wmma(const _Float16* __restrict__ h1, const float* __restrict__ w2,
                             const float* __restrict__ cb, const float* __restrict__ g,
                             const float* __restrict__ bb, _Float16* __restrict__ h2) {
    __shared__ _Float16 Bt[9][32][32];   // Bt[tap][n][c] = w2[n][c][tap]
    __shared__ _Float16 A[64][32];
    int tid = threadIdx.x, b = blockIdx.y;
    int pbase = blockIdx.x * 64;
    for (int idx = tid; idx < 9 * 32 * 32; idx += 256) {
        int tap = idx / 1024, r = idx & 1023, n = r >> 5, c = r & 31;
        Bt[tap][n][c] = (_Float16)w2[(n * 32 + c) * 9 + tap];
    }
    int lane = tid & 31, wave = tid >> 5;
    int mt = wave >> 1, nt = wave & 1;
    int hi = lane >> 4, lr = lane & 15;
    // staging coords for this thread (one v8h per tap)
    int spl = tid >> 2, sc = (tid & 3) * 8;
    int sp  = pbase + spl;
    int spc = sp < P2 ? sp : (P2 - 1);
    int sy  = spc / WW, sx = spc % WW;
    const v8h vzero = {};
    v8f acc = {};
    for (int tap = 0; tap < 9; ++tap) {
        int dy = tap / 3 - 1, dx = tap % 3 - 1;
        int yy = sy + dy, xc = sx + dx;
        bool ok = (sp < P2) & (yy >= 0) & (yy < HH) & (xc >= 0) & (xc < WW);
        int psrc = ok ? (spc + dy * WW + dx) : spc;        // always a valid pixel
        v8h val = *(const v8h*)(h1 + (((size_t)b * P2 + psrc) << 5) + sc);
        *(v8h*)&A[spl][sc] = ok ? val : vzero;
        __syncthreads();
        const v8h* ar = (const v8h*)&A[mt * 16 + lr][0];
        v16h af = SHUF16(ar[hi], ar[2 + hi]);            // K = e + (e>=8?8:0) + hi*8
        const v8h* br = (const v8h*)&Bt[tap][nt * 16 + lr][0];
        v16h bf = SHUF16(br[hi], br[2 + hi]);            // K = e + (e>=8?8:0) + hi*8
        acc = wmma16x16x32(af, bf, acc);
        __syncthreads();
    }
    int n = nt * 16 + lr;
    float c0 = cb[n], scale = g[n] * rsqrtf(1.f + EPSV), sh = bb[n];
#pragma unroll
    for (int r = 0; r < 8; ++r) {
        int ml = mt * 16 + r + hi * 8;
        int p = pbase + ml;
        if (p < P2) {
            float v = (acc[r] + c0) * scale + sh;
            v = v > 0.f ? v : 0.f;
            h2[(((size_t)b * P2 + p) << 5) + n] = (_Float16)v;
        }
    }
}

// ---------------------------------------------------------------- maxpool2x2 -> nodesh[:,20:4116] f16, zero K tail
__global__ void k_nodes_spec(const _Float16* __restrict__ h2, _Float16* __restrict__ nodesh) {
    int i = blockIdx.x, b = blockIdx.y, tid = threadIdx.x;
    const size_t rowp = (size_t)WW << 5;
    size_t row = (size_t)(b * NODES + i) * KPAD1;
    for (int k = tid; k < 4096; k += 256) {
        int t = i >> 1;
        int c = ((i & 1) << 4) + (k >> 8);
        int fq = k & 255;
        size_t base = ((size_t)(b * HH + fq * 2) * WW + t * 2) << 5;
        float v0 = (float)h2[base + c];
        float v1 = (float)h2[base + 32 + c];
        float v2 = (float)h2[base + rowp + c];
        float v3 = (float)h2[base + rowp + 32 + c];
        nodesh[row + 20 + k] = (_Float16)fmaxf(fmaxf(v0, v1), fmaxf(v2, v3));
    }
    if (tid < KPAD1 - NDIM) nodesh[row + NDIM + tid] = (_Float16)0.f;   // zero K padding
}

// ---------------------------------------------------------------- pack weights: (K x N) f32 -> Bt[N][Kpad] f16
__global__ void k_w_pack_t(const float* __restrict__ W, _Float16* __restrict__ Bt,
                           int K, int N, int Kpad) {
    int idx = blockIdx.x * 256 + threadIdx.x;
    if (idx >= N * Kpad) return;
    int n = idx / Kpad, k = idx % Kpad;
    Bt[(size_t)n * Kpad + k] = (k < K) ? (_Float16)W[(size_t)k * N + n] : (_Float16)0.f;
}

// ---------------------------------------------------------------- WMMA GEMM, fragments direct from global
// A: f16 row-major [>= gridDim.x*64 rows][lda], K-tail zero-padded.
// Bt: f16 [N][ldb] (B transposed), K-tail zero-padded.
// Per wave: one 16x16 C tile; fragment loads are contiguous b128s.
__global__ void k_gemm_wmma_h(const _Float16* __restrict__ Ag, int lda,
                              const _Float16* __restrict__ Btg, int ldb,
                              float* __restrict__ Cg, int M, int N, int K) {
    int tid = threadIdx.x;
    int lane = tid & 31, wave = tid >> 5;
    int mt = wave >> 1, nt = wave & 1, hi = lane >> 4, lr = lane & 15;
    int m0 = blockIdx.x * 64, n0 = blockIdx.y * 32;
    int m = m0 + mt * 16 + lr;
    int n = n0 + nt * 16 + lr;
    const _Float16* ap = Ag + (size_t)m * lda + hi * 8;
    const _Float16* bp = Btg + (size_t)n * ldb + hi * 8;
    v8f acc = {};
    for (int k0 = 0; k0 < K; k0 += 32) {
        v16h af = SHUF16(*(const v8h*)(ap), *(const v8h*)(ap + 16));  // K = e+(e>=8?8:0)+hi*8
        v16h bf = SHUF16(*(const v8h*)(bp), *(const v8h*)(bp + 16));  // K = e+(e>=8?8:0)+hi*8
        __builtin_prefetch(ap + 128, 0, 1);
        acc = wmma16x16x32(af, bf, acc);
        ap += 32; bp += 32;
    }
#pragma unroll
    for (int r = 0; r < 8; ++r) {
        int mm = m0 + mt * 16 + r + hi * 8;
        if (mm < M && n < N) Cg[(size_t)mm * N + n] = acc[r];
    }
}

// ---------------------------------------------------------------- GAT attention scores
__global__ void k_gat_scores(const float* __restrict__ h, const float* __restrict__ asrc,
                             const float* __restrict__ adst, float* __restrict__ ssrc,
                             float* __restrict__ sdst, int heads, int fo) {
    int id = blockIdx.x * 256 + threadIdx.x;
    if (id >= BB * NODES * heads) return;
    int hd = id % heads, row = id / heads;
    const float* hp = h + (size_t)row * heads * fo + hd * fo;
    float a = 0.f, d = 0.f;
    for (int f = 0; f < fo; ++f) { float v = hp[f]; a += v * asrc[hd * fo + f]; d += v * adst[hd * fo + f]; }
    ssrc[id] = a; sdst[id] = d;
}

// ---------------------------------------------------------------- dense GAT attention (softmax over 250)
// grid (250, B, heads); block = fo threads. HOUT: write f16 (feeds next GEMM).
template <bool HOUT>
__global__ void k_gat_attn(const float* __restrict__ h, const float* __restrict__ ssrc,
                           const float* __restrict__ sdst, const float* __restrict__ bias,
                           float* __restrict__ outf, _Float16* __restrict__ outh,
                           int heads, int fo, int do_relu) {
    __shared__ float w[NODES];
    __shared__ float red[64];
    int i = blockIdx.x, b = blockIdx.y, hd = blockIdx.z, tid = threadIdx.x;
    int bd = blockDim.x;
    float sd = sdst[(b * NODES + i) * heads + hd];
    float lmax = -1e30f;
    for (int j = tid; j < NODES; j += bd) {
        float e = sd + ssrc[(b * NODES + j) * heads + hd];
        e = e > 0.f ? e : 0.2f * e;                 // leaky_relu 0.2
        w[j] = e;
        lmax = fmaxf(lmax, e);
    }
    red[tid] = lmax; __syncthreads();
    for (int s = bd >> 1; s > 0; s >>= 1) { if (tid < s) red[tid] = fmaxf(red[tid], red[tid + s]); __syncthreads(); }
    float mx = red[0]; __syncthreads();
    float lsum = 0.f;
    for (int j = tid; j < NODES; j += bd) { float e = __expf(w[j] - mx); w[j] = e; lsum += e; }
    red[tid] = lsum; __syncthreads();
    for (int s = bd >> 1; s > 0; s >>= 1) { if (tid < s) red[tid] += red[tid + s]; __syncthreads(); }
    float inv = 1.f / red[0];
    __syncthreads();
    float acc = 0.f;
    const float* hc = h + hd * fo + tid;
    for (int j = 0; j < NODES; ++j)
        acc += w[j] * hc[(size_t)(b * NODES + j) * heads * fo];
    acc = acc * inv + bias[hd * fo + tid];
    if (do_relu) acc = acc > 0.f ? acc : 0.f;
    size_t ofs = (size_t)(b * NODES + i) * heads * fo + hd * fo + tid;
    if (HOUT) outh[ofs] = (_Float16)acc;
    else      outf[ofs] = acc;
}

// ---------------------------------------------------------------- mean over nodes + MLP head
__global__ void k_head(const float* __restrict__ g2, const float* __restrict__ fc1w,
                       const float* __restrict__ fc1b, const float* __restrict__ bnfg,
                       const float* __restrict__ bnfb, const float* __restrict__ fc2w,
                       const float* __restrict__ fc2b, float* __restrict__ dout) {
    __shared__ float emb[32];
    __shared__ float z[128];
    int b = blockIdx.x, t = threadIdx.x;
    if (t < 32) {
        float a = 0.f;
        for (int i = 0; i < NODES; ++i) a += g2[(size_t)(b * NODES + i) * 32 + t];
        a *= (1.f / (float)NODES);
        emb[t] = a;
        dout[16 + b * 32 + t] = a;               // emb output (after 8x2 logits)
    }
    __syncthreads();
    float acc = fc1b[t];
    for (int k = 0; k < 32; ++k) acc += emb[k] * fc1w[k * 128 + t];
    acc = acc * (bnfg[t] * rsqrtf(1.f + EPSV)) + bnfb[t];
    z[t] = acc > 0.f ? acc : 0.f;
    __syncthreads();
    if (t < 2) {
        float a = fc2b[t];
        for (int k = 0; k < 128; ++k) a += z[k] * fc2w[k * 2 + t];
        dout[b * 2 + t] = a;
    }
}

// ================================================================ launch
extern "C" void kernel_launch(void* const* d_in, const int* in_sizes, int n_in,
                              void* d_out, int out_size, void* d_ws, size_t ws_size,
                              hipStream_t stream) {
    (void)in_sizes; (void)n_in; (void)out_size; (void)ws_size;
    const float* x       = (const float*)d_in[0];
    const float* sinc_w  = (const float*)d_in[1];
    const float* conv1_w = (const float*)d_in[2];
    const float* conv1_b = (const float*)d_in[3];
    const float* bn1_g   = (const float*)d_in[4];
    const float* bn1_b   = (const float*)d_in[5];
    const float* conv2_w = (const float*)d_in[6];
    const float* conv2_b = (const float*)d_in[7];
    const float* bn2_g   = (const float*)d_in[8];
    const float* bn2_b   = (const float*)d_in[9];
    const float* gat1_W  = (const float*)d_in[10];
    const float* gat1_as = (const float*)d_in[11];
    const float* gat1_ad = (const float*)d_in[12];
    const float* gat1_bi = (const float*)d_in[13];
    const float* gat2_W  = (const float*)d_in[14];
    const float* gat2_as = (const float*)d_in[15];
    const float* gat2_ad = (const float*)d_in[16];
    const float* gat2_bi = (const float*)d_in[17];
    const float* fc1_w   = (const float*)d_in[18];
    const float* fc1_b   = (const float*)d_in[19];
    const float* bnf_g   = (const float*)d_in[20];
    const float* bnf_b   = (const float*)d_in[21];
    const float* fc2_w   = (const float*)d_in[22];
    const float* fc2_b   = (const float*)d_in[23];
    float* dout = (float*)d_out;

    char* wsp = (char*)d_ws;
    auto alloc = [&](size_t bytes) -> void* {
        void* p = (void*)wsp;
        wsp += (bytes + 255) & ~(size_t)255;
        return p;
    };
    float*    sbuf   = (float*)   alloc((size_t)BB * HH * WW * 4);           // logmag
    _Float16* h1     = (_Float16*)alloc((size_t)BB * P2 * 32 * 2);           // NHWC f16
    _Float16* h2     = (_Float16*)alloc((size_t)BB * P2 * 32 * 2);           // NHWC f16
    _Float16* nodesh = (_Float16*)alloc((size_t)MPAD * KPAD1 * 2);           // padded f16 A for GEMM1
    _Float16* w1t    = (_Float16*)alloc((size_t)512 * KPAD1 * 2);            // Bt for GEMM1
    float*    h1g    = (float*)   alloc((size_t)BB * NODES * 512 * 4);
    float*    ssrc1  = (float*)   alloc((size_t)BB * NODES * 8 * 4);
    float*    sdst1  = (float*)   alloc((size_t)BB * NODES * 8 * 4);
    _Float16* g1h    = (_Float16*)alloc((size_t)MPAD * 512 * 2);             // padded f16 A for GEMM2
    _Float16* w2t    = (_Float16*)alloc((size_t)32 * 512 * 2);               // Bt for GEMM2
    float*    h2g    = (float*)   alloc((size_t)BB * NODES * 32 * 4);
    float*    ssrc2  = (float*)   alloc((size_t)BB * NODES * 4);
    float*    sdst2  = (float*)   alloc((size_t)BB * NODES * 4);
    float*    g2     = (float*)   alloc((size_t)BB * NODES * 32 * 4);

    k_fft_logmag<<<dim3(WW, BB), 128, 0, stream>>>(x, sbuf);
    k_sinc_nodes<<<(BB * NODES * 20 + 255) / 256, 256, 0, stream>>>(x, sinc_w, nodesh);
    k_conv1<<<dim3((P2 + 255) / 256, BB), 256, 0, stream>>>(sbuf, conv1_w, conv1_b, bn1_g, bn1_b, h1);
    k_conv2_wmma<<<dim3((P2 + 63) / 64, BB), 256, 0, stream>>>(h1, conv2_w, conv2_b, bn2_g, bn2_b, h2);
    k_nodes_spec<<<dim3(NODES, BB), 256, 0, stream>>>(h2, nodesh);
    k_w_pack_t<<<(512 * KPAD1 + 255) / 256, 256, 0, stream>>>(gat1_W, w1t, NDIM, 512, KPAD1);
    k_w_pack_t<<<(32 * 512 + 255) / 256, 256, 0, stream>>>(gat2_W, w2t, 512, 32, 512);

    // GAT1 projection: (2000 x 4128) @ (4128 x 512)
    k_gemm_wmma_h<<<dim3(MPAD / 64, 512 / 32), 256, 0, stream>>>(nodesh, KPAD1, w1t, KPAD1, h1g, BB * NODES, 512, KPAD1);
    k_gat_scores<<<(BB * NODES * 8 + 255) / 256, 256, 0, stream>>>(h1g, gat1_as, gat1_ad, ssrc1, sdst1, 8, 64);
    k_gat_attn<true><<<dim3(NODES, BB, 8), 64, 0, stream>>>(h1g, ssrc1, sdst1, gat1_bi, nullptr, g1h, 8, 64, 1);

    // GAT2 projection: (2000 x 512) @ (512 x 32)
    k_gemm_wmma_h<<<dim3(MPAD / 64, 1), 256, 0, stream>>>(g1h, 512, w2t, 512, h2g, BB * NODES, 32, 512);
    k_gat_scores<<<(BB * NODES + 255) / 256, 256, 0, stream>>>(h2g, gat2_as, gat2_ad, ssrc2, sdst2, 1, 32);
    k_gat_attn<false><<<dim3(NODES, BB, 1), 32, 0, stream>>>(h2g, ssrc2, sdst2, gat2_bi, g2, nullptr, 1, 32, 0);

    k_head<<<BB, 128, 0, stream>>>(g2, fc1_w, fc1_b, bnf_g, bnf_b, fc2_w, fc2_b, dout);
}
